// HYV3MoEFused_53970559042167
// MI455X (gfx1250) — compile-verified, matrix operand order checked by reading, not run on hardware
//
#include <hip/hip_runtime.h>
#include <hip/hip_bf16.h>

#define HDIM 2048
#define IDIM 1024
#define NEXP 16
#define TOPK 4
#define KB   32      // K-step per LDS stage
#define LDST 40      // LDS row stride in bf16 elems (32 + 8 pad) = 80B, 16B aligned

typedef __attribute__((ext_vector_type(8)))  __bf16 v8bf;
typedef __attribute__((ext_vector_type(16))) __bf16 v16bf;
typedef __attribute__((ext_vector_type(8)))  float  v8f;

union FragU { v16bf f; v8bf h[2]; };
union Cvt16 { v8bf v[2]; __bf16 e[16]; };

// ---- CDNA5 async data mover: global -> LDS, 16B per lane, tracked by ASYNCcnt
__device__ __forceinline__ void asyncLoadLdsB128(unsigned ldsOff, const void* gaddr) {
  asm volatile("global_load_async_to_lds_b128 %0, %1, off"
               :: "v"(ldsOff), "v"(gaddr)
               : "memory");
}
__device__ __forceinline__ void waitAsync0() {
  asm volatile("s_wait_asynccnt 0x0" ::: "memory");
}
// low 32 bits of a generic pointer to __shared__ == LDS byte offset on AMDGPU
__device__ __forceinline__ unsigned ldsOffsetOf(const void* p) {
  return (unsigned)(size_t)p;
}

// Load a 16x32 bf16 WMMA operand fragment from a row-major LDS tile.
// Per ISA 7.12.2: lane<16 -> K in {0..7,16..23}; lane>=16 -> K in {8..15,24..31}.
__device__ __forceinline__ v16bf ldsFrag(const __bf16* base, int row, int kb) {
  FragU u;
  const __bf16* p = base + row * LDST + kb;
  u.h[0] = *(const v8bf*)(p);
  u.h[1] = *(const v8bf*)(p + 16);
  return u.f;
}

// ---------------- router: sigmoid gating, biased top-4 selection, renormalized raw scores
__global__ void moe_router(const float* __restrict__ x, const float* __restrict__ gw,
                           const float* __restrict__ bias,
                           int* __restrict__ tidx, float* __restrict__ tw) {
  const int t   = blockIdx.x;
  const int tid = threadIdx.x;            // 128 threads
  __shared__ float red[NEXP][128];
  float part[NEXP];
#pragma unroll
  for (int e = 0; e < NEXP; e++) part[e] = 0.f;
  for (int j = tid; j < HDIM; j += 128) {
    float xv = x[(size_t)t * HDIM + j];
#pragma unroll
    for (int e = 0; e < NEXP; e++) part[e] += xv * gw[e * HDIM + j];
  }
#pragma unroll
  for (int e = 0; e < NEXP; e++) red[e][tid] = part[e];
  __syncthreads();
  for (int off = 64; off > 0; off >>= 1) {
    if (tid < off)
      for (int e = 0; e < NEXP; e++) red[e][tid] += red[e][tid + off];
    __syncthreads();
  }
  if (tid == 0) {
    float sc[NEXP], bi[NEXP];
    for (int e = 0; e < NEXP; e++) {
      float s = 1.f / (1.f + __expf(-red[e][0]));
      sc[e] = s;
      bi[e] = s + bias[e];
    }
    float sum = 0.f; int idx[TOPK]; float wv[TOPK];
    for (int k = 0; k < TOPK; k++) {
      int b = 0; float bv = -3.4e38f;
      for (int e = 0; e < NEXP; e++)
        if (bi[e] > bv) { bv = bi[e]; b = e; }
      idx[k] = b; wv[k] = sc[b]; bi[b] = -3.4e38f; sum += wv[k];
    }
    float inv = 1.f / sum;
    for (int k = 0; k < TOPK; k++) {
      tidx[t * TOPK + k] = idx[k];
      tw[t * TOPK + k]   = wv[k] * inv;
    }
  }
}

// ---------------- fp32 -> bf16 activation conversion
__global__ void moe_cvt_bf16(const float* __restrict__ x, __bf16* __restrict__ xb, int n) {
  int i = (blockIdx.x * 256 + threadIdx.x) * 8;
  if (i + 8 <= n) {
    v8bf o;
#pragma unroll
    for (int j = 0; j < 8; j++) o[j] = (__bf16)x[i + j];
    *(v8bf*)(xb + i) = o;
  }
}

// ---------------- deterministic per-expert compaction (single block)
__global__ void moe_compact(const int* __restrict__ tidx, const float* __restrict__ tw,
                            int* __restrict__ rowTok, float* __restrict__ rowW,
                            int* __restrict__ offs, int T) {
  __shared__ int cnt[NEXP];
  __shared__ int off[NEXP + 1];
  const int e = threadIdx.x;
  const int n = T * TOPK;
  if (e < NEXP) {
    int c = 0;
    for (int j = 0; j < n; j++) if (tidx[j] == e) c++;
    cnt[e] = c;
  }
  __syncthreads();
  if (e == 0) {
    int a = 0;
    for (int i = 0; i < NEXP; i++) { off[i] = a; a += cnt[i]; }
    off[NEXP] = a;
    for (int i = 0; i <= NEXP; i++) offs[i] = off[i];
  }
  __syncthreads();
  if (e < NEXP) {
    int p = off[e];
    for (int j = 0; j < n; j++)
      if (tidx[j] == e) { rowTok[p] = j >> 2; rowW[p] = tw[j]; p++; }
  }
}

// ---------------- gate/up grouped GEMM + SiLU*mul -> bf16 inter
// mode=1: routed (gather rows via rowTok, expert = blockIdx.x, bounds via offs)
// mode=0: shared  (rows are tokens directly)
__global__ void __launch_bounds__(256) moe_gateup(
    const __bf16* __restrict__ xb,
    const float* __restrict__ wgAll, const float* __restrict__ wuAll,
    const int* __restrict__ rowTok, const int* __restrict__ offs,
    __bf16* __restrict__ interOut, int mode, int T) {
  __shared__ __bf16 lA[64 * LDST];
  __shared__ __bf16 lG[128 * LDST];
  __shared__ __bf16 lU[128 * LDST];

  const int tid = threadIdx.x;
  int rowBase, rowEnd;
  const float *wg, *wu;
  if (mode) {
    const int e = blockIdx.x;
    rowEnd  = offs[e + 1];
    rowBase = offs[e] + blockIdx.y * 64;
    if (rowBase >= rowEnd) return;
    wg = wgAll + (size_t)e * IDIM * HDIM;
    wu = wuAll + (size_t)e * IDIM * HDIM;
  } else {
    rowBase = blockIdx.y * 64; rowEnd = T;
    wg = wgAll; wu = wuAll;
  }
  const int i0 = blockIdx.z * 128;

  // A-stage identity: 4 threads per row, 8 bf16 (16B) each -> async to LDS
  const int arow = tid >> 2, aseg = tid & 3;
  int r = rowBase + arow; if (r > rowEnd - 1) r = rowEnd - 1;
  const int gtok = mode ? rowTok[r] : r;
  const __bf16* aSrc = xb + (size_t)gtok * HDIM + aseg * 8;
  const unsigned aDst = ldsOffsetOf(&lA[arow * LDST + aseg * 8]);

  // B-stage identity: 2 threads per row, 16 fp32 -> 16 bf16 each
  const int brow = tid >> 1, bhalf = tid & 1;
  const float* gSrc = wg + (size_t)(i0 + brow) * HDIM + bhalf * 16;
  const float* uSrc = wu + (size_t)(i0 + brow) * HDIM + bhalf * 16;

  // wave identity: 8 waves, 4(M) x 2(N), wave tile 16x64
  const int lane = tid & 31, wid = tid >> 5;
  const int wm = wid & 3, wn = wid >> 2;
  const int m16 = lane & 15;
  const int kb = (lane < 16) ? 0 : 8;

  v8f cg[4] = {}; v8f cu[4] = {};

  for (int k0 = 0; k0 < HDIM; k0 += KB) {
    __syncthreads();                           // previous tile fully consumed
    asyncLoadLdsB128(aDst, aSrc + k0);         // activations: async global->LDS
    {
      const float* p = gSrc + k0;
      Cvt16 c;
#pragma unroll
      for (int j = 0; j < 16; j++) c.e[j] = (__bf16)p[j];
      *(v8bf*)(&lG[brow * LDST + bhalf * 16])     = c.v[0];
      *(v8bf*)(&lG[brow * LDST + bhalf * 16 + 8]) = c.v[1];
    }
    {
      const float* p = uSrc + k0;
      Cvt16 c;
#pragma unroll
      for (int j = 0; j < 16; j++) c.e[j] = (__bf16)p[j];
      *(v8bf*)(&lU[brow * LDST + bhalf * 16])     = c.v[0];
      *(v8bf*)(&lU[brow * LDST + bhalf * 16 + 8]) = c.v[1];
    }
    if (k0 + KB < HDIM) {                      // prefetch next weight tiles
      __builtin_prefetch((const void*)(gSrc + k0 + KB), 0, 0);
      __builtin_prefetch((const void*)(uSrc + k0 + KB), 0, 0);
    }
    waitAsync0();                              // ASYNCcnt == 0: LDS A-tile landed
    __syncthreads();

    v16bf a = ldsFrag(lA, wm * 16 + m16, kb);
#pragma unroll
    for (int s = 0; s < 4; s++) {
      v16bf bg = ldsFrag(lG, wn * 64 + s * 16 + m16, kb);
      cg[s] = __builtin_amdgcn_wmma_f32_16x16x32_bf16(false, a, false, bg, (short)0, cg[s], false, false);
      v16bf bu = ldsFrag(lU, wn * 64 + s * 16 + m16, kb);
      cu[s] = __builtin_amdgcn_wmma_f32_16x16x32_bf16(false, a, false, bu, (short)0, cu[s], false, false);
    }
  }

  // epilogue: SiLU(g)*u -> inter[r][i], masked for partial tiles
  const int mhi = (lane >> 4) << 3;     // C/D layout: lanes 16-31 hold M=8..15
#pragma unroll
  for (int s = 0; s < 4; s++) {
#pragma unroll
    for (int v = 0; v < 8; v++) {
      int rr = rowBase + wm * 16 + v + mhi;
      if (rr < rowEnd) {
        float g = cg[s][v], u = cu[s][v];
        float val = g * __builtin_amdgcn_rcpf(1.f + __expf(-g)) * u;
        interOut[(size_t)rr * IDIM + i0 + wn * 64 + s * 16 + m16] = (__bf16)val;
      }
    }
  }
}

// ---------------- down GEMM: out[token] (+)= w * (inter_row @ wd^T)
// mode=1: routed expert e (accumulate, weight rowW); mode=0: shared (store, weight 1)
__global__ void __launch_bounds__(256) moe_down(
    const __bf16* __restrict__ interBuf, const float* __restrict__ wd,
    float* __restrict__ out,
    const int* __restrict__ rowTok, const float* __restrict__ rowW,
    const int* __restrict__ offs, int e, int mode, int T) {
  __shared__ __bf16 lA[64 * LDST];
  __shared__ __bf16 lB[128 * LDST];
  const int tid = threadIdx.x;
  int rowBase, rowEnd;
  if (mode) {
    rowEnd  = offs[e + 1];
    rowBase = offs[e] + blockIdx.x * 64;
    if (rowBase >= rowEnd) return;
  } else {
    rowBase = blockIdx.x * 64; rowEnd = T;
  }
  const int h0 = blockIdx.y * 128;

  const int arow = tid >> 2, aseg = tid & 3;
  int r = rowBase + arow; if (r > rowEnd - 1) r = rowEnd - 1;
  const __bf16* aSrc = interBuf + (size_t)r * IDIM + aseg * 8;
  const unsigned aDst = ldsOffsetOf(&lA[arow * LDST + aseg * 8]);

  const int brow = tid >> 1, bhalf = tid & 1;
  const float* bSrc = wd + (size_t)(h0 + brow) * IDIM + bhalf * 16;

  const int lane = tid & 31, wid = tid >> 5;
  const int wm = wid & 3, wn = wid >> 2;
  const int m16 = lane & 15;
  const int kb = (lane < 16) ? 0 : 8;

  v8f cd[4] = {};
  for (int k0 = 0; k0 < IDIM; k0 += KB) {
    __syncthreads();
    asyncLoadLdsB128(aDst, aSrc + k0);         // bf16 inter rows: async global->LDS
    {
      const float* p = bSrc + k0;
      Cvt16 c;
#pragma unroll
      for (int j = 0; j < 16; j++) c.e[j] = (__bf16)p[j];
      *(v8bf*)(&lB[brow * LDST + bhalf * 16])     = c.v[0];
      *(v8bf*)(&lB[brow * LDST + bhalf * 16 + 8]) = c.v[1];
    }
    if (k0 + KB < IDIM)
      __builtin_prefetch((const void*)(bSrc + k0 + KB), 0, 0);
    waitAsync0();
    __syncthreads();
    v16bf a = ldsFrag(lA, wm * 16 + m16, kb);
#pragma unroll
    for (int s = 0; s < 4; s++) {
      v16bf b = ldsFrag(lB, wn * 64 + s * 16 + m16, kb);
      cd[s] = __builtin_amdgcn_wmma_f32_16x16x32_bf16(false, a, false, b, (short)0, cd[s], false, false);
    }
  }

  const int mhi = (lane >> 4) << 3;
#pragma unroll
  for (int s = 0; s < 4; s++) {
#pragma unroll
    for (int v = 0; v < 8; v++) {
      int rr = rowBase + wm * 16 + v + mhi;
      if (rr < rowEnd) {
        int tok   = mode ? rowTok[rr] : rr;
        float wgt = mode ? rowW[rr] : 1.f;
        size_t oi = (size_t)tok * HDIM + h0 + wn * 64 + s * 16 + m16;
        float val = wgt * cd[s][v];
        if (mode) out[oi] = out[oi] + val;   // unique (tok,h) per launch -> race-free
        else      out[oi] = val;             // shared expert initializes out
      }
    }
  }
}

extern "C" void kernel_launch(void* const* d_in, const int* in_sizes, int n_in,
                              void* d_out, int out_size, void* d_ws, size_t ws_size,
                              hipStream_t stream) {
  const float* x       = (const float*)d_in[0];
  const float* gate_w  = (const float*)d_in[1];
  const float* ebias   = (const float*)d_in[2];
  const float* w_gate  = (const float*)d_in[3];
  const float* w_up    = (const float*)d_in[4];
  const float* w_down  = (const float*)d_in[5];
  const float* sw_gate = (const float*)d_in[6];
  const float* sw_up   = (const float*)d_in[7];
  const float* sw_down = (const float*)d_in[8];
  float* out = (float*)d_out;
  const int T = in_sizes[0] / HDIM;   // 2048

  char* w = (char*)d_ws;
  auto alloc = [&](size_t bytes) -> char* {
    char* p = w;
    w += (bytes + 255) & ~(size_t)255;
    return p;
  };
  __bf16* xb     = (__bf16*)alloc((size_t)T * HDIM * 2);          //  8 MB
  __bf16* sinter = (__bf16*)alloc((size_t)T * IDIM * 2);          //  4 MB
  __bf16* inter  = (__bf16*)alloc((size_t)T * TOPK * IDIM * 2);   // 16 MB
  int*    tidx   = (int*)  alloc((size_t)T * TOPK * 4);
  float*  tw     = (float*)alloc((size_t)T * TOPK * 4);
  int*    rowTok = (int*)  alloc((size_t)T * TOPK * 4);
  float*  rowW   = (float*)alloc((size_t)T * TOPK * 4);
  int*    offs   = (int*)  alloc((NEXP + 1) * 4);

  moe_router<<<T, 128, 0, stream>>>(x, gate_w, ebias, tidx, tw);
  moe_cvt_bf16<<<(T * HDIM / 8 + 255) / 256, 256, 0, stream>>>(x, xb, T * HDIM);
  moe_compact<<<1, 32, 0, stream>>>(tidx, tw, rowTok, rowW, offs, T);

  dim3 gi(NEXP, (T + 63) / 64, IDIM / 128);
  moe_gateup<<<gi, 256, 0, stream>>>(xb, w_gate, w_up, rowTok, offs, inter, 1, T);
  dim3 gs(1, T / 64, IDIM / 128);
  moe_gateup<<<gs, 256, 0, stream>>>(xb, sw_gate, sw_up, nullptr, nullptr, sinter, 0, T);

  dim3 gd(T / 64, HDIM / 128);
  moe_down<<<gd, 256, 0, stream>>>(sinter, sw_down, out, nullptr, nullptr, nullptr, 0, 0, T);
  dim3 gr((T + 63) / 64, HDIM / 128);
  for (int e = 0; e < NEXP; e++)
    moe_down<<<gr, 256, 0, stream>>>(inter, w_down + (size_t)e * HDIM * IDIM, out,
                                     rowTok, rowW, offs, e, 1, T);
}
